// RefinementTFNLayer_31001073943170
// MI455X (gfx1250) — matrix operand order repaired and checked by reading, not compiled, not run on hardware
//
#include <hip/hip_runtime.h>
#include <hip/hip_bf16.h>
#include <math.h>

#define NS     16
#define NV     4
#define NG     50
#define HID    64
#define WNUM   400          // (NS+NV)*NS + (NS+NV)*NV = 320 + 80
#define IN_DIM 82           // NG + 2*NS
#define K1PAD  96           // 82 padded to 3*32
#define NATTR  28           // NS + 3*NV
#define TE     32           // edges per block (two WMMA M row-groups)
#define SQRT3F     1.7320508075688772f
#define INV_SQRT20 0.22360679774997896f // 1/sqrt(NS+NV)
#define GS_STOPF   5.0f

typedef __attribute__((ext_vector_type(16))) _Float16 v16h;
typedef __attribute__((ext_vector_type(8)))  float    v8f;

union Frag16 { v16h v; _Float16 h[16]; };

// ---------------------------------------------------------------------------
// Prep: convert W1 (82x64) and W2 (64x400) fp32 -> f16, pre-swizzled into the
// per-lane WMMA B-fragment layout so each lane loads 16 contiguous halves.
// B order used here (contiguous k, A loads the same): lane l -> col = l&15, K = kstep*32 + (l>>4)*16 + j
// ---------------------------------------------------------------------------
__global__ void prep_weights(const float* __restrict__ W1,
                             const float* __restrict__ W2,
                             _Float16* __restrict__ w1t,
                             _Float16* __restrict__ w2t) {
  int idx = blockIdx.x * blockDim.x + threadIdx.x;
  const int W1TOT = 4 * 3 * 32 * 16;   // 4 col-tiles x 3 k-steps
  if (idx < W1TOT) {
    int j    = idx & 15;
    int lane = (idx >> 4) & 31;
    int tile = idx >> 9;               // 0..11
    int ks   = tile % 3;
    int ct   = tile / 3;
    int col  = ct * 16 + (lane & 15);
    int k    = ks * 32 + ((lane >> 4) * 16) + j;
    float v  = (k < IN_DIM) ? W1[k * HID + col] : 0.0f;
    w1t[idx] = (_Float16)v;
  }
  const int W2TOT = 25 * 2 * 32 * 16;  // 25 col-tiles x 2 k-steps
  if (idx < W2TOT) {
    int j    = idx & 15;
    int lane = (idx >> 4) & 31;
    int tile = idx >> 9;               // 0..49
    int ks   = tile & 1;
    int t    = tile >> 1;
    int col  = t * 16 + (lane & 15);
    int k    = ks * 32 + ((lane >> 4) * 16) + j;
    w2t[idx] = (_Float16)W2[k * WNUM + col];
  }
}

__global__ void zero_ws(float* __restrict__ agg, float* __restrict__ cnt, int N) {
  int i = blockIdx.x * blockDim.x + threadIdx.x;
  if (i < N * NATTR) agg[i] = 0.0f;
  if (i < N)         cnt[i] = 0.0f;
}

// ---------------------------------------------------------------------------
// Fused per-edge kernel: featurize -> MLP via WMMA -> equivariant contraction
// folded tile-by-tile into a 28-feature message accumulator -> global scatter.
// 32 edges per block = 2 WMMA row-groups; 8 wave32 waves.
// ---------------------------------------------------------------------------
__global__ __launch_bounds__(256)
void tfn_edge_kernel(const float* __restrict__ node_attr,
                     const float* __restrict__ pos,
                     const float* __restrict__ b1,
                     const float* __restrict__ b2,
                     const int*   __restrict__ edge_index,
                     const _Float16* __restrict__ w1t,
                     const _Float16* __restrict__ w2t,
                     float* __restrict__ agg,
                     float* __restrict__ cnt,
                     int E) {
  __shared__ _Float16 efL[TE * K1PAD];     // 32x96 padded edge features (f16)
  __shared__ _Float16 hL [TE * HID];       // 32x64 hidden activations (f16)
  __shared__ float in0eL[TE * NS];         // src scalar channels
  __shared__ float in1oL[TE * NV * 3];     // src vector channels
  __shared__ float shL  [TE * 3];          // sqrt(3)*[uy,uz,ux]
  __shared__ float o0L  [TE * 20];         // scalar-path input vector
  __shared__ float msgL [TE * NATTR];      // message accumulator
  __shared__ float distL[TE];
  __shared__ int   srcL[TE], dstL[TE], validL[TE];

  const int tid  = threadIdx.x;
  const int lane = tid & 31;
  const int wv   = tid >> 5;
  const int n    = lane & 15;   // WMMA column / A-row index within row-group
  const int hi   = lane >> 4;   // lane half-group
  const int e0   = blockIdx.x * TE;

  // ---- phase 0a: indices + geometry -------------------------------------
  if (tid < TE) {
    int e = e0 + tid;
    int valid = (e < E) ? 1 : 0;
    int s = valid ? edge_index[e]     : 0;
    int d = valid ? edge_index[E + e] : 0;
    srcL[tid] = s; dstL[tid] = d; validL[tid] = valid;
    float dx = pos[d * 3 + 0] - pos[s * 3 + 0];
    float dy = pos[d * 3 + 1] - pos[s * 3 + 1];
    float dz = pos[d * 3 + 2] - pos[s * 3 + 2];
    float dist = sqrtf(dx * dx + dy * dy + dz * dz);
    distL[tid] = dist;
    float inv = 1.0f / fmaxf(dist, 1e-8f);
    shL[tid * 3 + 0] = SQRT3F * dy * inv;
    shL[tid * 3 + 1] = SQRT3F * dz * inv;
    shL[tid * 3 + 2] = SQRT3F * dx * inv;
  }
  __syncthreads();

  // ---- phase 0b: gather node features, RBFs, padding, accum init --------
  for (int idx = tid; idx < TE * NATTR; idx += 256) {
    int m = idx / NATTR, k = idx % NATTR;
    float v = validL[m] ? node_attr[srcL[m] * NATTR + k] : 0.0f;
    if (k < NS) { in0eL[m * NS + k] = v; efL[m * K1PAD + NG + k] = (_Float16)v; }
    else        { in1oL[m * 12 + (k - NS)] = v; }
  }
  for (int idx = tid; idx < TE * NS; idx += 256) {
    int m = idx / NS, k = idx % NS;
    float v = validL[m] ? node_attr[dstL[m] * NATTR + k] : 0.0f;
    efL[m * K1PAD + NG + NS + k] = (_Float16)v;
  }
  const float step  = GS_STOPF / (float)(NG - 1);
  const float coeff = -0.5f / (step * step);
  for (int idx = tid; idx < TE * NG; idx += 256) {
    int m = idx / NG, g = idx % NG;
    float x = distL[m] - step * (float)g + 1e-6f;
    efL[m * K1PAD + g] = (_Float16)__expf(coeff * x * x);
  }
  for (int idx = tid; idx < TE * (K1PAD - IN_DIM); idx += 256) {
    int m = idx / (K1PAD - IN_DIM), k = idx % (K1PAD - IN_DIM);
    efL[m * K1PAD + IN_DIM + k] = (_Float16)0.0f;
  }
  for (int idx = tid; idx < TE * NATTR; idx += 256) msgL[idx] = 0.0f;
  __syncthreads();

  // ---- o0 vector (needs in0e/in1o/sh complete) --------------------------
  for (int idx = tid; idx < TE * 20; idx += 256) {
    int m = idx / 20, i = idx % 20;
    float v;
    if (i < NS) v = in0eL[m * NS + i];
    else {
      int vi = i - NS;
      v = (in1oL[m * 12 + vi * 3 + 0] * shL[m * 3 + 0] +
           in1oL[m * 12 + vi * 3 + 1] * shL[m * 3 + 1] +
           in1oL[m * 12 + vi * 3 + 2] * shL[m * 3 + 2]) * (1.0f / SQRT3F);
    }
    o0L[idx] = v;
  }
  __syncthreads();

  // ---- GEMM1: h = relu(ef @ W1 + b1) ------------------------------------
  // 8 tiles = 2 row-groups x 4 col-tiles -> one per wave (all waves busy)
  {
    const int rg = wv >> 2;          // row-group (edge block of 16)
    const int ct = wv & 3;           // column tile
    v8f c = {0.f, 0.f, 0.f, 0.f, 0.f, 0.f, 0.f, 0.f};
    #pragma unroll
    for (int ks = 0; ks < 3; ++ks) {
      Frag16 a, b;
      // A layout: lane -> row (rg*16+n); K chunk [hi*16..+15]
      const _Float16* ab = &efL[(rg * 16 + n) * K1PAD + ks * 32 + hi * 16];
      #pragma unroll
      for (int j = 0; j < 16; ++j) a.h[j] = ab[j];
      const _Float16* bb = &w1t[((ct * 3 + ks) * 32 + lane) * 16];
      #pragma unroll
      for (int j = 0; j < 16; ++j) b.h[j] = bb[j];
      c = __builtin_amdgcn_wmma_f32_16x16x32_f16(false, a.v, false, b.v,
                                                 (short)0, c, false, false);
    }
    float bias = b1[ct * 16 + n];
    #pragma unroll
    for (int r = 0; r < 8; ++r) {
      int m = rg * 16 + r + hi * 8;
      hL[m * HID + ct * 16 + n] = (_Float16)fmaxf(c[r] + bias, 0.0f);
    }
  }
  __syncthreads();

  // ---- GEMM2 (h @ W2 + b2) fused with the equivariant contraction -------
  // 50 tiles = 2 row-groups x 25 col-tiles, round-robin over 8 waves.
  // Col-tile t<20: cols = i*16+o with i=t  -> out_0e[o] += o0[i] * w
  // Col-tile t>=20: cols = i*4+o (i=(t-20)*4 + n/4, o=n%4)
  //                 -> out_1o[o,c] += o1[i,c] * w
  for (int tt = wv; tt < 50; tt += 8) {
    const int rg = tt / 25;
    const int t  = tt % 25;
    v8f c = {0.f, 0.f, 0.f, 0.f, 0.f, 0.f, 0.f, 0.f};
    #pragma unroll
    for (int ks = 0; ks < 2; ++ks) {
      Frag16 a, b;
      const _Float16* ab = &hL[(rg * 16 + n) * HID + ks * 32 + hi * 16];
      #pragma unroll
      for (int j = 0; j < 16; ++j) a.h[j] = ab[j];
      const _Float16* bb = &w2t[((t * 2 + ks) * 32 + lane) * 16];
      #pragma unroll
      for (int j = 0; j < 16; ++j) b.h[j] = bb[j];
      c = __builtin_amdgcn_wmma_f32_16x16x32_f16(false, a.v, false, b.v,
                                                 (short)0, c, false, false);
    }
    float bias = b2[t * 16 + n];
    if (t < 20) {
      #pragma unroll
      for (int r = 0; r < 8; ++r) {
        int m = rg * 16 + r + hi * 8;
        float val = (c[r] + bias) * INV_SQRT20;
        atomicAdd(&msgL[m * NATTR + n], o0L[m * 20 + t] * val);
      }
    } else {
      int tp = t - 20;
      int i  = tp * 4 + (n >> 2);
      int o  = n & 3;
      #pragma unroll
      for (int r = 0; r < 8; ++r) {
        int m = rg * 16 + r + hi * 8;
        float val = (c[r] + bias) * INV_SQRT20;
        #pragma unroll
        for (int cd = 0; cd < 3; ++cd) {
          float o1v = (i < NS) ? in0eL[m * NS + i] * shL[m * 3 + cd]
                               : in1oL[m * 12 + (i - NS) * 3 + cd];
          atomicAdd(&msgL[m * NATTR + NS + o * 3 + cd], o1v * val);
        }
      }
    }
  }
  __syncthreads();

  // ---- scatter messages + counts to nodes -------------------------------
  for (int idx = tid; idx < TE * NATTR; idx += 256) {
    int m = idx / NATTR, f = idx % NATTR;
    if (validL[m]) atomicAdd(&agg[dstL[m] * NATTR + f], msgL[idx]);
  }
  if (tid < TE && validL[tid]) atomicAdd(&cnt[dstL[tid]], 1.0f);
}

__global__ void finalize_kernel(const float* __restrict__ node_attr,
                                const float* __restrict__ agg,
                                const float* __restrict__ cnt,
                                float* __restrict__ out, int N) {
  int i = blockIdx.x * blockDim.x + threadIdx.x;
  if (i < N * NATTR) {
    int nd = i / NATTR;
    out[i] = node_attr[i] + agg[i] / fmaxf(cnt[nd], 1.0f);
  }
}

extern "C" void kernel_launch(void* const* d_in, const int* in_sizes, int n_in,
                              void* d_out, int out_size, void* d_ws, size_t ws_size,
                              hipStream_t stream) {
  const float* node_attr  = (const float*)d_in[0];
  const float* pos        = (const float*)d_in[1];
  const float* W1         = (const float*)d_in[2];
  const float* b1         = (const float*)d_in[3];
  const float* W2         = (const float*)d_in[4];
  const float* b2         = (const float*)d_in[5];
  const int*   edge_index = (const int*)d_in[6];

  const int N = in_sizes[0] / NATTR;
  const int E = in_sizes[6] / 2;
  float* out = (float*)d_out;

  char* ws = (char*)d_ws;
  _Float16* w1t = (_Float16*)ws;                    // 12 KB swizzled W1 (f16)
  _Float16* w2t = (_Float16*)(ws + 16 * 1024);      // 50 KB swizzled W2 (f16)
  float* agg = (float*)(ws + 80 * 1024);            // N*28 floats
  float* cnt = agg + (size_t)N * NATTR;             // N floats

  prep_weights<<<(25 * 2 * 32 * 16 + 255) / 256, 256, 0, stream>>>(W1, W2, w1t, w2t);
  zero_ws<<<(N * NATTR + 255) / 256, 256, 0, stream>>>(agg, cnt, N);
  tfn_edge_kernel<<<(E + TE - 1) / TE, 256, 0, stream>>>(node_attr, pos, b1, b2,
                                                         edge_index, w1t, w2t,
                                                         agg, cnt, E);
  finalize_kernel<<<(N * NATTR + 255) / 256, 256, 0, stream>>>(node_attr, agg, cnt, out, N);
}
